// FlaxLlamaAttention_10617159156228
// MI455X (gfx1250) — matrix operand order repaired and checked by reading, not compiled, not run on hardware
//
#include <hip/hip_runtime.h>
#include <hip/hip_bf16.h>
#include <math.h>

// ---------------------------------------------------------------------------
// Types
// ---------------------------------------------------------------------------
typedef __bf16 bf16;
typedef __bf16 v2bf  __attribute__((ext_vector_type(2)));
typedef __bf16 v4bf  __attribute__((ext_vector_type(4)));
typedef __bf16 v8bf  __attribute__((ext_vector_type(8)));
typedef __bf16 v16bf __attribute__((ext_vector_type(16)));
typedef float  v8f   __attribute__((ext_vector_type(8)));
typedef float  v4f   __attribute__((ext_vector_type(4)));
typedef unsigned int u32x4 __attribute__((ext_vector_type(4)));
typedef unsigned int u32x8 __attribute__((ext_vector_type(8)));
typedef int          i32x4 __attribute__((ext_vector_type(4)));
typedef int          i32x8 __attribute__((ext_vector_type(8)));

#define HIDDEN   4096
#define SEQ      2048
#define NH       32
#define NKV      8
#define HD       128

#define HAVE_TDM (__has_builtin(__builtin_amdgcn_tensor_load_to_lds) && \
                  __has_builtin(__builtin_amdgcn_s_wait_tensorcnt))

// f32 -> bf16 RNE: let the backend pick the native convert (v_cvt_*bf16*)
__device__ __forceinline__ bf16 f2bf(float f) { return (bf16)f; }

__device__ __forceinline__ v2bf f2bf2(float a, float b) {
#if __has_builtin(__builtin_amdgcn_cvt_pk_bf16_f32)
    return __builtin_bit_cast(v2bf, __builtin_amdgcn_cvt_pk_bf16_f32(a, b));
#else
    v2bf r; r[0] = f2bf(a); r[1] = f2bf(b); return r;
#endif
}

__device__ __forceinline__ v8bf pack8(v4f x0, v4f x1) {
    v2bf p0 = f2bf2(x0[0], x0[1]);
    v2bf p1 = f2bf2(x0[2], x0[3]);
    v2bf p2 = f2bf2(x1[0], x1[1]);
    v2bf p3 = f2bf2(x1[2], x1[3]);
    v4bf q0 = __builtin_shufflevector(p0, p1, 0, 1, 2, 3);
    v4bf q1 = __builtin_shufflevector(p2, p3, 0, 1, 2, 3);
    return __builtin_shufflevector(q0, q1, 0, 1, 2, 3, 4, 5, 6, 7);
}

// A-operand fragment (16 rows x 32 K): lane<16 -> K {0..7,16..23}, lane>=16 -> K {8..15,24..31}
__device__ __forceinline__ v16bf load_frag_a(const bf16* __restrict__ base, int ld, int lane) {
    int r  = lane & 15;
    int k0 = (lane < 16) ? 0 : 8;
    const bf16* p = base + (size_t)r * ld + k0;
    v8bf lo = *(const v8bf*)(p);
    v8bf hi = *(const v8bf*)(p + 16);
    return __builtin_shufflevector(lo, hi, 0,1,2,3,4,5,6,7,8,9,10,11,12,13,14,15);
}

// B-operand fragment (32 K x 16 cols): lane<16 holds K {0..7,16..23} of column n=lane, lane>=16 K {8..15,24..31}.
// `base` is stored row-per-N (K contiguous within a lane's row).
__device__ __forceinline__ v16bf load_frag_b(const bf16* __restrict__ base, int ld, int lane) {
    int n  = lane & 15;
    int k0 = (lane < 16) ? 0 : 8;
    const bf16* p = base + (size_t)n * ld + k0;
    v8bf lo = *(const v8bf*)(p);
    v8bf hi = *(const v8bf*)(p + 16);
    return __builtin_shufflevector(lo, hi, 0,1,2,3,4,5,6,7,8,9,10,11,12,13,14,15);
}

__device__ __forceinline__ v8f wmma_bf16(v16bf a, v16bf b, v8f c) {
    return __builtin_amdgcn_wmma_f32_16x16x32_bf16(false, a, false, b, (short)0, c, false, false);
}

// ---------------------------------------------------------------------------
// Tensor Data Mover: 2D tile load (bf16 elements) global -> LDS.
// D# per CDNA5 ISA. Dims/strides in elements, addresses bytes.
// ---------------------------------------------------------------------------
__device__ __forceinline__ void tdm_load_2d(unsigned lds_addr, const void* gaddr,
                                            unsigned tensor_d0, unsigned tensor_d1,
                                            unsigned tile_d0, unsigned tile_d1,
                                            unsigned stride0) {
#if HAVE_TDM
    unsigned long long ga = (unsigned long long)(size_t)gaddr;
    u32x4 g0;
    g0[0] = 1u;                                     // count=1, user desc
    g0[1] = lds_addr;                               // lds_addr [63:32]
    g0[2] = (unsigned)ga;                           // global_addr [95:64]
    g0[3] = (unsigned)(ga >> 32) | (2u << 30);      // ga[56:32] | type=2 ("image")
    u32x8 g1;
    g1[0] = (1u << 16);                             // data_size = 1 -> 2 bytes
    g1[1] = (tensor_d0 & 0xFFFFu) << 16;            // tensor_dim0[15:0] @ [63:48]
    g1[2] = (tensor_d0 >> 16) | ((tensor_d1 & 0xFFFFu) << 16);
    g1[3] = (tensor_d1 >> 16) | ((tile_d0 & 0xFFFFu) << 16);
    g1[4] = tile_d1 & 0xFFFFu;                      // tile_dim1; tile_dim2 = 0
    g1[5] = stride0;                                // tensor_dim0_stride[31:0]
    g1[6] = 0u;
    g1[7] = 0u;
    i32x4 z4 = {};
#if __clang_major__ >= 23
    i32x8 z8 = {};
    __builtin_amdgcn_tensor_load_to_lds(g0, __builtin_bit_cast(i32x8, g1), z4, z4, z8, 0);
#else
    __builtin_amdgcn_tensor_load_to_lds(g0, __builtin_bit_cast(i32x8, g1), z4, z4, 0);
#endif
#else
    (void)lds_addr; (void)gaddr; (void)tensor_d0; (void)tensor_d1;
    (void)tile_d0; (void)tile_d1; (void)stride0;
#endif
}

// ---------------------------------------------------------------------------
// GEMM: C[M,N](f32) = A[M,K](f32) * B[K,N](f32); bf16 WMMA, f32 accumulate.
// 256 threads (8 waves, 4x2 wave grid). BM=128, BN=128, KB=32.
// Double-buffered LDS, software-pipelined global fetch, 1 barrier per K-step.
// ---------------------------------------------------------------------------
#define BM 128
#define BN 128
#define KB 32

__global__ __launch_bounds__(256)
void gemm_f32_bf16(const float* __restrict__ A, const float* __restrict__ B,
                   float* __restrict__ C, int M, int N, int K) {
    __shared__ __align__(16) bf16 As[2][BM * KB];   // 2 x 8 KB, row-major [m][k]
    __shared__ __align__(16) bf16 Bs[2][BN * KB];   // 2 x 8 KB, transposed [n][k]

    const int t    = threadIdx.x;
    const int lane = t & 31;
    const int wave = t >> 5;
    const int wy   = wave & 3;     // 4 M sub-tiles of 32 rows
    const int wx   = wave >> 2;    // 2 N halves of 64 cols
    const int m0   = blockIdx.y * BM;
    const int n0   = blockIdx.x * BN;

    const int ar = t >> 1,  ac = (t & 1) * 16;   // A: 128 rows x 32 cols, 16 f32/thread
    const int bk = t >> 3,  bc = (t & 7) * 16;   // B: 32 rows x 128 cols, 16 f32/thread
    const float* ga = A + (size_t)(m0 + ar) * K + ac;
    const float* gb = B + (size_t)bk * N + n0 + bc;

    v4f a_reg[4], b_reg[4];
    const int nkb = K / KB;

    auto fetch = [&](int kb) {
        const v4f* pa = (const v4f*)(ga + (size_t)kb * KB);
        a_reg[0] = pa[0]; a_reg[1] = pa[1]; a_reg[2] = pa[2]; a_reg[3] = pa[3];
        const v4f* pb = (const v4f*)(gb + (size_t)kb * KB * N);
        b_reg[0] = pb[0]; b_reg[1] = pb[1]; b_reg[2] = pb[2]; b_reg[3] = pb[3];
        if (kb + 1 < nkb) {
            __builtin_prefetch(ga + (size_t)(kb + 1) * KB, 0, 1);
            __builtin_prefetch(gb + (size_t)(kb + 1) * KB * N, 0, 1);
        }
    };
    auto stage = [&](int buf) {
        *(v8bf*)(&As[buf][ar * KB + ac])     = pack8(a_reg[0], a_reg[1]);
        *(v8bf*)(&As[buf][ar * KB + ac + 8]) = pack8(a_reg[2], a_reg[3]);
        #pragma unroll
        for (int j = 0; j < 16; j += 2) {    // transpose B into [n][k]
            v2bf p = f2bf2(b_reg[j >> 2][j & 3], b_reg[(j + 1) >> 2][(j + 1) & 3]);
            Bs[buf][(bc + j)     * KB + bk] = p[0];
            Bs[buf][(bc + j + 1) * KB + bk] = p[1];
        }
    };

    v8f acc[2][4] = {};
    fetch(0);
    for (int kb = 0; kb < nkb; ++kb) {
        const int buf = kb & 1;
        stage(buf);
        __syncthreads();
        if (kb + 1 < nkb) fetch(kb + 1);     // overlap next global fetch with WMMAs

        v16bf af0 = load_frag_a(&As[buf][(wy * 32)      * KB], KB, lane);
        v16bf af1 = load_frag_a(&As[buf][(wy * 32 + 16) * KB], KB, lane);
        #pragma unroll
        for (int n = 0; n < 4; ++n) {
            v16bf bfg = load_frag_b(&Bs[buf][(wx * 64 + n * 16) * KB], KB, lane);
            acc[0][n] = wmma_bf16(af0, bfg, acc[0][n]);
            acc[1][n] = wmma_bf16(af1, bfg, acc[1][n]);
        }
    }

    const int rr = (lane < 16) ? 0 : 8;
    const int cn = lane & 15;
    #pragma unroll
    for (int mt = 0; mt < 2; ++mt)
        #pragma unroll
        for (int n = 0; n < 4; ++n)
            #pragma unroll
            for (int r = 0; r < 8; ++r) {
                int row = m0 + wy * 32 + mt * 16 + r + rr;
                int col = n0 + wx * 64 + n * 16 + cn;
                C[(size_t)row * N + col] = acc[mt][n][r];
            }
}

// ---------------------------------------------------------------------------
// RoPE + layout prep.
// ---------------------------------------------------------------------------
__global__ __launch_bounds__(256)
void rope_prep(const float* __restrict__ Qraw, const float* __restrict__ Kraw,
               const float* __restrict__ Vraw, const int* __restrict__ pos_ids,
               bf16* __restrict__ Qr, bf16* __restrict__ Kr, bf16* __restrict__ Vt) {
    const int p = blockIdx.x;
    const float pos = (float)pos_ids[p];
    const float l2t_64 = 13.287712379549449f / 64.0f;   // log2(10000)/64

    for (int idx = threadIdx.x; idx < NH * HD; idx += 256) {
        int h = idx >> 7, d = idx & 127;
        float freq = exp2f(-(float)(d & 63) * l2t_64);
        float s, c; __sincosf(pos * freq, &s, &c);
        float x  = Qraw[(size_t)p * (NH * HD) + idx];
        float xr = (d < 64) ? -Qraw[(size_t)p * (NH * HD) + h * HD + d + 64]
                            :  Qraw[(size_t)p * (NH * HD) + h * HD + d - 64];
        Qr[((size_t)h * SEQ + p) * HD + d] = f2bf(x * c + xr * s);
    }
    for (int idx = threadIdx.x; idx < NKV * HD; idx += 256) {
        int h = idx >> 7, d = idx & 127;
        float freq = exp2f(-(float)(d & 63) * l2t_64);
        float s, c; __sincosf(pos * freq, &s, &c);
        float x  = Kraw[(size_t)p * (NKV * HD) + idx];
        float xr = (d < 64) ? -Kraw[(size_t)p * (NKV * HD) + h * HD + d + 64]
                            :  Kraw[(size_t)p * (NKV * HD) + h * HD + d - 64];
        Kr[((size_t)h * SEQ + p) * HD + d] = f2bf(x * c + xr * s);
        Vt[((size_t)h * HD + d) * SEQ + p] = f2bf(Vraw[(size_t)p * (NKV * HD) + idx]);
    }
}

// ---------------------------------------------------------------------------
// Causal GQA flash attention. Grid (16 q-blocks, 32 heads), 256 threads.
// K/V tiles double-buffered in LDS, fed by the Tensor Data Mover pipelined one
// key-block ahead: wave 0 issues the next tile's two tensor loads right after
// the publish barrier, s_wait_tensorcnt retires them at the top of the next
// iteration, so the DMA overlaps the 16 WMMAs + softmax of the current step.
// ---------------------------------------------------------------------------
#define QB   128
#define KBLK 32

__global__ __launch_bounds__(256)
void attn_kernel(const bf16* __restrict__ Q,    // [NH][SEQ][HD]
                 const bf16* __restrict__ Kc,   // [NKV][SEQ][HD]
                 const bf16* __restrict__ Vt,   // [NKV][HD][SEQ]
                 float* __restrict__ O) {       // [SEQ][NH*HD]
    __shared__ __align__(16) bf16 Ks[2][KBLK * HD];   // 2 x 8 KB  [key][dim]
    __shared__ __align__(16) bf16 Vs[2][HD * KBLK];   // 2 x 8 KB  [dim][key]
    __shared__ __align__(16) bf16 Pt[8][16 * KBLK];   // 8 KB per-wave P re-layout

    const int h    = blockIdx.y;
    const int kvh  = h >> 2;                      // GQA: 4 Q heads per KV head
    const int qb   = blockIdx.x;
    const int lane = threadIdx.x & 31;
    const int wave = threadIdx.x >> 5;
    const int q0   = qb * QB + wave * 16;

    const bf16* Qh = Q  + (size_t)h   * SEQ * HD;
    const bf16* Kh = Kc + (size_t)kvh * SEQ * HD;
    const bf16* Vh = Vt + (size_t)kvh * HD * SEQ;
    const unsigned ldsK[2] = { (unsigned)(size_t)&Ks[0][0], (unsigned)(size_t)&Ks[1][0] };
    const unsigned ldsV[2] = { (unsigned)(size_t)&Vs[0][0], (unsigned)(size_t)&Vs[1][0] };

    // preload this wave's Q tile as 4 A-fragments (head dim = 4 chunks of 32)
    v16bf qf[4];
    #pragma unroll
    for (int c = 0; c < 4; ++c)
        qf[c] = load_frag_a(Qh + (size_t)q0 * HD + c * 32, HD, lane);

    v8f o[8] = {};
    float m_i[8], l_i[8];
    #pragma unroll
    for (int r = 0; r < 8; ++r) { m_i[r] = -INFINITY; l_i[r] = 0.0f; }

    const int   qhalf = (lane < 16) ? 0 : 8;
    const int   cn    = lane & 15;
    const float scale = 0.08838834764831845f;     // 1/sqrt(128)
    const int   nkb   = (qb + 1) * (QB / KBLK);   // causal: keys up to block end

#if HAVE_TDM
    if (wave == 0) {   // prologue: kick tile 0 into buffer 0
        tdm_load_2d(ldsK[0], Kh, HD, SEQ, HD, KBLK, HD);
        tdm_load_2d(ldsV[0], Vh, SEQ, HD, KBLK, HD, SEQ);
    }
#endif

    for (int kb = 0; kb < nkb; ++kb) {
        const int k0 = kb * KBLK;
        const int p  = kb & 1;

#if HAVE_TDM
        if (wave == 0) __builtin_amdgcn_s_wait_tensorcnt(0);   // tile kb landed
#else
        {
            const v8bf* srcK = (const v8bf*)(Kh + (size_t)k0 * HD);
            v8bf* dstK = (v8bf*)&Ks[p][0];
            for (int i = threadIdx.x; i < KBLK * HD / 8; i += 256) dstK[i] = srcK[i];
            for (int i = threadIdx.x; i < HD * KBLK / 8; i += 256) {
                int d = i >> 2, c = (i & 3) * 8;
                *(v8bf*)(&Vs[p][d * KBLK + c]) = *(const v8bf*)(Vh + (size_t)d * SEQ + k0 + c);
            }
        }
#endif
        __syncthreads();                          // publish buffer p
#if HAVE_TDM
        if (wave == 0 && kb + 1 < nkb) {          // pipeline: kick tile kb+1
            tdm_load_2d(ldsK[p ^ 1], Kh + (size_t)(k0 + KBLK) * HD, HD, SEQ, HD, KBLK, HD);
            tdm_load_2d(ldsV[p ^ 1], Vh + (k0 + KBLK), SEQ, HD, KBLK, HD, SEQ);
        }
#endif

        // ---- S = Q * K^T : two 16-key tiles, 4 hd-chunks each ----
        v8f s[2] = {};
        #pragma unroll
        for (int nt = 0; nt < 2; ++nt) {
            #pragma unroll
            for (int c = 0; c < 4; ++c) {
                v16bf kf = load_frag_b(&Ks[p][(size_t)(nt * 16) * HD + c * 32], HD, lane);
                s[nt] = wmma_bf16(qf[c], kf, s[nt]);
            }
        }

        // ---- scale, causal mask, online softmax ----
        float mx[8];
        #pragma unroll
        for (int r = 0; r < 8; ++r) {
            int qrow = q0 + r + qhalf;
            #pragma unroll
            for (int nt = 0; nt < 2; ++nt) {
                int krow = k0 + nt * 16 + cn;
                float v = s[nt][r] * scale;
                if (krow > qrow) v = -INFINITY;
                s[nt][r] = v;
            }
            float m = fmaxf(s[0][r], s[1][r]);
            #pragma unroll
            for (int off = 1; off < 16; off <<= 1)
                m = fmaxf(m, __shfl_xor(m, off, 32));
            mx[r] = m;
        }
        float alpha[8];
        #pragma unroll
        for (int r = 0; r < 8; ++r) {
            float mn = fmaxf(m_i[r], mx[r]);
            alpha[r] = __expf(m_i[r] - mn);
            m_i[r] = mn;
            float ps = 0.0f;
            #pragma unroll
            for (int nt = 0; nt < 2; ++nt) {
                float pv = __expf(s[nt][r] - mn);
                s[nt][r] = pv;
                ps += pv;
            }
            #pragma unroll
            for (int off = 1; off < 16; off <<= 1)
                ps += __shfl_xor(ps, off, 32);
            l_i[r] = l_i[r] * alpha[r] + ps;
        }
        #pragma unroll
        for (int n = 0; n < 8; ++n)
            #pragma unroll
            for (int r = 0; r < 8; ++r)
                o[n][r] *= alpha[r];

        // ---- re-layout P (C-fragment -> A-fragment) via per-wave LDS tile ----
        #pragma unroll
        for (int nt = 0; nt < 2; ++nt)
            #pragma unroll
            for (int r = 0; r < 8; ++r)
                Pt[wave][(r + qhalf) * KBLK + nt * 16 + cn] = f2bf(s[nt][r]);
        __syncthreads();                          // publish Pt
        v16bf pf = load_frag_a(&Pt[wave][0], KBLK, lane);

        // ---- O += P * V : 8 dim-tiles ----
        #pragma unroll
        for (int n = 0; n < 8; ++n) {
            v16bf vf = load_frag_b(&Vs[p][(size_t)(n * 16) * KBLK], KBLK, lane);
            o[n] = wmma_bf16(pf, vf, o[n]);
        }
        __syncthreads();                          // all done reading buffer p
    }

    // ---- finalize: O /= l, store f32 in [seq][h*128+d] for the out-proj GEMM ----
    #pragma unroll
    for (int r = 0; r < 8; ++r) {
        float inv = 1.0f / l_i[r];
        int row = q0 + r + qhalf;
        #pragma unroll
        for (int n = 0; n < 8; ++n)
            O[(size_t)row * (NH * HD) + h * HD + n * 16 + cn] = o[n][r] * inv;
    }
}

// ---------------------------------------------------------------------------
// Launcher. Inputs: hidden_states, position_ids, wq, wk, wv, wo
// Workspace (~76 MB): Qraw(32M) Kraw(8M) Vraw(8M) Qr(16M) Kr(4M) Vt(4M);
// attention output reuses the Qraw buffer.
// ---------------------------------------------------------------------------
extern "C" void kernel_launch(void* const* d_in, const int* in_sizes, int n_in,
                              void* d_out, int out_size, void* d_ws, size_t ws_size,
                              hipStream_t stream) {
    (void)in_sizes; (void)n_in; (void)out_size; (void)ws_size;
    const float* hidden = (const float*)d_in[0];
    const int*   posids = (const int*)d_in[1];
    const float* wq     = (const float*)d_in[2];
    const float* wk     = (const float*)d_in[3];
    const float* wv     = (const float*)d_in[4];
    const float* wo     = (const float*)d_in[5];
    float* out = (float*)d_out;

    char* ws = (char*)d_ws;
    float* Qraw = (float*)ws; ws += (size_t)SEQ * HIDDEN * 4;
    float* Kraw = (float*)ws; ws += (size_t)SEQ * NKV * HD * 4;
    float* Vraw = (float*)ws; ws += (size_t)SEQ * NKV * HD * 4;
    bf16*  Qr   = (bf16*)ws;  ws += (size_t)NH  * SEQ * HD * 2;
    bf16*  Kr   = (bf16*)ws;  ws += (size_t)NKV * SEQ * HD * 2;
    bf16*  Vt   = (bf16*)ws;  ws += (size_t)NKV * HD * SEQ * 2;
    float* attn = Qraw;   // Qraw dead after rope_prep

    dim3 blk(256);
    gemm_f32_bf16<<<dim3(HIDDEN / BN,     SEQ / BM), blk, 0, stream>>>(hidden, wq, Qraw, SEQ, HIDDEN, HIDDEN);
    gemm_f32_bf16<<<dim3(NKV * HD / BN,   SEQ / BM), blk, 0, stream>>>(hidden, wk, Kraw, SEQ, NKV * HD, HIDDEN);
    gemm_f32_bf16<<<dim3(NKV * HD / BN,   SEQ / BM), blk, 0, stream>>>(hidden, wv, Vraw, SEQ, NKV * HD, HIDDEN);
    rope_prep<<<dim3(SEQ), blk, 0, stream>>>(Qraw, Kraw, Vraw, posids, Qr, Kr, Vt);
    attn_kernel<<<dim3(SEQ / QB, NH), blk, 0, stream>>>(Qr, Kr, Vt, attn);
    gemm_f32_bf16<<<dim3(HIDDEN / BN,     SEQ / BM), blk, 0, stream>>>(attn, wo, out, SEQ, HIDDEN, HIDDEN);
}
